// Wasserstein2Loss_67757404062369
// MI455X (gfx1250) — matrix operand order repaired and checked by reading, not compiled, not run on hardware
//
#include <hip/hip_runtime.h>

// Sinkhorn / entropic Wasserstein-2 loss for (4,128,128) x (4,128,128) -> (4,4)
// One workgroup per (b0,b1) pair; log-domain Gaussian blur implemented as
// exp-domain GEMM on the f16 WMMA pipe (v_wmma_f32_16x16x32_f16), with
// per-row max subtraction for stability. phi buffers live in d_ws.
// The Gaussian kernel matrix K (symmetric Toeplitz) is precomputed once in
// LDS as f16, so B fragments are pure ds_load_b128 traffic.

#define NN      128
#define ESTR    136          // E stride in halves (padded; 272 B = 16*17 -> 16B-aligned rows)
#define PSTR    132          // P stride in floats (padded -> conflict-free transposed stores)
#define W2EPS   1e-3f
#define W2ITER  10

typedef __attribute__((ext_vector_type(16))) _Float16 v16h;
typedef __attribute__((ext_vector_type(8)))  _Float16 v8h;
typedef __attribute__((ext_vector_type(8)))  float    v8f;

struct __align__(16) SinkSmem {
    _Float16 E[NN * ESTR];   // 34816 B : exp-normalized A operand (f16)
    _Float16 KF[NN * NN];    // 32768 B : Gaussian kernel, row-major (symmetric -> B operand)
    float    P[NN * PSTR];   // 67584 B : transposed pass-1 result (f32)
    float    Ms[NN];         //   512 B : per-row maxima
    float    red[4];         //    16 B : [0]=sum p, [1]=sum phi0*a0, [2]=sum phi1*a1
};

__device__ __forceinline__ v16h cat8(v8h lo, v8h hi) {
    return __builtin_shufflevector(lo, hi, 0,1,2,3,4,5,6,7,8,9,10,11,12,13,14,15);
}

// One full separable 2D log-blur of phiIn (global, 128x128 row-major).
// MODE 0: phiOut[o] = log(max(aEpi[o],1e-20)) - blur2d(phiIn)[o]
// MODE 1: red[0]   += sum_o exp(phiEpi[o] + blur2d(phiIn)[o])
template <int MODE>
__device__ void blur2d(SinkSmem& sm,
                       const float* __restrict__ phiIn,
                       float* __restrict__ phiOut,
                       const float* __restrict__ aEpi,
                       const float* __restrict__ phiEpi)
{
    const int tid  = threadIdx.x;
    const int lane = tid & 31;
    const int wave = tid >> 5;
    const int mr   = lane & 15;      // M-row / N-col sub-index within tile
    const int hi   = lane >> 4;      // half-wave selector
    const int m0   = wave * 16;      // this wave's 16-row strip

    __syncthreads();                 // phiIn (global) / KF produced by this block

    // ---------- pass 1: blur along the last axis; result stored transposed in P
    if (tid < NN) {
        const float* r = phiIn + tid * NN;
        float m = -3.0e38f;
        for (int i = 0; i < NN; ++i) m = fmaxf(m, r[i]);
        sm.Ms[tid] = m;
    }
    __syncthreads();
    for (int idx = tid; idx < NN * NN; idx += blockDim.x) {
        int r = idx >> 7, c = idx & (NN - 1);
        sm.E[r * ESTR + c] = (_Float16)__expf(phiIn[r * NN + c] - sm.Ms[r]);
    }
    __syncthreads();
    {
        v8f acc[8];
#pragma unroll
        for (int nb = 0; nb < 8; ++nb) { v8f z = {}; acc[nb] = z; }
        for (int kb = 0; kb < 4; ++kb) {
            // A fragment: 16x32 f16 (lanes 0-15: K 0-7/16-23, lanes 16-31: K 8-15/24-31)
            const v8h* ap = (const v8h*)&sm.E[(m0 + mr) * ESTR + kb * 32 + hi * 8];
            v16h a = cat8(ap[0], ap[2]);
#pragma unroll
            for (int nb = 0; nb < 8; ++nb) {
                const int ncol = nb * 16 + mr;
                // B fragment: 32x16 f16; K symmetric -> row slice (lanes 0-15: K 0-7/16-23, lanes 16-31: K 8-15/24-31)
                const v8h* bp = (const v8h*)&sm.KF[ncol * NN + kb * 32 + hi * 8];
                v16h b = cat8(bp[0], bp[2]);
                acc[nb] = __builtin_amdgcn_wmma_f32_16x16x32_f16(
                    false, a, false, b, (short)0, acc[nb], false, false);
            }
        }
#pragma unroll
        for (int nb = 0; nb < 8; ++nb) {
            const int ncol = nb * 16 + mr;
#pragma unroll
            for (int v = 0; v < 8; ++v) {
                int row = m0 + v + hi * 8;   // C/D layout: lanes 0-15 -> M=v, 16-31 -> M=8+v
                sm.P[ncol * PSTR + row] = __logf(fmaxf(acc[nb][v], 1e-30f)) + sm.Ms[row];
            }
        }
    }
    __syncthreads();

    // ---------- pass 2: blur along the other axis (rows of P); epilogue per MODE
    if (tid < NN) {
        const float* r = &sm.P[tid * PSTR];
        float m = -3.0e38f;
        for (int i = 0; i < NN; ++i) m = fmaxf(m, r[i]);
        sm.Ms[tid] = m;
    }
    __syncthreads();
    for (int idx = tid; idx < NN * NN; idx += blockDim.x) {
        int r = idx >> 7, c = idx & (NN - 1);
        sm.E[r * ESTR + c] = (_Float16)__expf(sm.P[r * PSTR + c] - sm.Ms[r]);
    }
    __syncthreads();
    {
        float psum = 0.0f;
        v8f acc[8];
#pragma unroll
        for (int nb = 0; nb < 8; ++nb) { v8f z = {}; acc[nb] = z; }
        for (int kb = 0; kb < 4; ++kb) {
            const v8h* ap = (const v8h*)&sm.E[(m0 + mr) * ESTR + kb * 32 + hi * 8];
            v16h a = cat8(ap[0], ap[2]);
#pragma unroll
            for (int nb = 0; nb < 8; ++nb) {
                const int ncol = nb * 16 + mr;
                const v8h* bp = (const v8h*)&sm.KF[ncol * NN + kb * 32 + hi * 8];
                v16h b = cat8(bp[0], bp[2]);
                acc[nb] = __builtin_amdgcn_wmma_f32_16x16x32_f16(
                    false, a, false, b, (short)0, acc[nb], false, false);
            }
        }
#pragma unroll
        for (int nb = 0; nb < 8; ++nb) {
            const int ncol = nb * 16 + mr;
#pragma unroll
            for (int v = 0; v < 8; ++v) {
                int row = m0 + v + hi * 8;
                float val = __logf(fmaxf(acc[nb][v], 1e-30f)) + sm.Ms[row];
                int o = ncol * NN + row;     // double transpose -> natural orientation
                if (MODE == 0) {
                    phiOut[o] = __logf(fmaxf(aEpi[o], 1e-20f)) - val;
                } else {
                    psum += __expf(phiEpi[o] + val);
                }
            }
        }
        if (MODE == 1) atomicAdd(&sm.red[0], psum);   // ds_add_f32
    }
    __syncthreads();
}

__global__ __launch_bounds__(256)
void Wasserstein2Loss_67757404062369_kernel(const float* __restrict__ a0,
                                            const float* __restrict__ a1,
                                            float* __restrict__ out,
                                            float* __restrict__ ws)
{
    __shared__ SinkSmem sm;
    const int pair = blockIdx.x;          // 0..15, row-major (b0,b1)
    const int b0 = pair >> 2, b1 = pair & 3;
    const float* A0 = a0 + (size_t)b0 * NN * NN;
    const float* A1 = a1 + (size_t)b1 * NN * NN;
    float* phi0 = ws + (size_t)pair * 2 * NN * NN;
    float* phi1 = phi0 + NN * NN;
    const int tid = threadIdx.x;
    const float invb = 1.0f / (W2EPS * (float)(NN * NN));   // 1/b, b = eps/pixel^2

    if (tid == 0) { sm.red[0] = 0.0f; sm.red[1] = 0.0f; sm.red[2] = 0.0f; }
    // Precompute the Gaussian kernel matrix once (row-major; symmetric).
    for (int idx = tid; idx < NN * NN; idx += blockDim.x) {
        int n = idx >> 7, k = idx & (NN - 1);
        float d = (float)(k - n);
        sm.KF[idx] = (_Float16)__expf(-d * d * invb);
        phi1[idx] = 0.0f;
    }
    // blur2d() begins with __syncthreads(), ordering the init above.

    for (int it = 0; it < W2ITER; ++it) {
        blur2d<0>(sm, phi1, phi0, A0, nullptr);   // phi0 = la0 - blur2d(phi1)
        blur2d<0>(sm, phi0, phi1, A1, nullptr);   // phi1 = la1 - blur2d(phi0)
    }

    // marginal terms: eps * (sum(phi0*a0) + sum(phi1*a1))
    float s0 = 0.0f, s1 = 0.0f;
    for (int idx = tid; idx < NN * NN; idx += blockDim.x) {
        s0 += phi0[idx] * A0[idx];
        s1 += phi1[idx] * A1[idx];
    }
    atomicAdd(&sm.red[1], s0);
    atomicAdd(&sm.red[2], s1);

    // transport-plan mass: sum exp(phi0 + blur2d(phi1))
    blur2d<1>(sm, phi1, nullptr, nullptr, phi0);

    if (tid == 0)
        out[pair] = -W2EPS * sm.red[0] + W2EPS * (sm.red[1] + sm.red[2]);
}

extern "C" void kernel_launch(void* const* d_in, const int* in_sizes, int n_in,
                              void* d_out, int out_size, void* d_ws, size_t ws_size,
                              hipStream_t stream)
{
    (void)in_sizes; (void)n_in; (void)out_size; (void)ws_size;
    const float* a0 = (const float*)d_in[0];
    const float* a1 = (const float*)d_in[1];
    float* out = (float*)d_out;
    float* ws  = (float*)d_ws;     // needs 16 * 2 * 128*128 * 4B = 2 MB
    hipLaunchKernelGGL(Wasserstein2Loss_67757404062369_kernel,
                       dim3(16), dim3(256), 0, stream, a0, a1, out, ws);
}
